// NodeInference_2138893713783
// MI455X (gfx1250) — matrix operand order repaired and checked by reading, not compiled, not run on hardware
//
#include <hip/hip_runtime.h>
#include <hip/hip_bf16.h>
#include <math.h>

typedef __attribute__((ext_vector_type(8)))  _Float16 v8h;
typedef __attribute__((ext_vector_type(16))) _Float16 v16h;
typedef __attribute__((ext_vector_type(8)))  float    v8f;

#define IN_FEATS 128
#define HEADS 8
#define HID 16
#define KTOT 8
#define MDIM 64
#define GCOLS 512

// ---------- helpers ----------
__device__ __forceinline__ unsigned enc_f(float f) {
    unsigned u = __float_as_uint(f);
    return (u & 0x80000000u) ? ~u : (u | 0x80000000u);
}
__device__ __forceinline__ float dec_f(unsigned k) {
    return (k & 0x80000000u) ? __uint_as_float(k ^ 0x80000000u) : __uint_as_float(~k);
}
__device__ __forceinline__ float lrelu(float x, float s) { return x > 0.f ? x : x * s; }

__device__ __forceinline__ v16h combine8(v8h lo, v8h hi) {
    v16h r;
    #pragma unroll
    for (int q = 0; q < 8; q++) { r[q] = lo[q]; r[q + 8] = hi[q]; }
    return r;
}

// ---------- transpose + f32->f16 convert (weights: src [rows x cols] -> dst [cols x rows]) ----------
__global__ void k_transpose_f16(const float* __restrict__ src, _Float16* __restrict__ dst,
                                int rows, int cols) {
    int i = blockIdx.x * blockDim.x + threadIdx.x;
    if (i >= rows * cols) return;
    int c = i / rows, r = i % rows;
    dst[(size_t)c * rows + r] = (_Float16)src[(size_t)r * cols + c];
}

// ---------- WMMA GEMM: C[n x ncols] = A[n x 128] (f32, optional leaky0.01) @ B (Bt f16 [ncols x 128]) ----------
__global__ __launch_bounds__(256) void k_gemm_wmma(const float* __restrict__ A,
                                                   const _Float16* __restrict__ Bt,
                                                   float* __restrict__ C,
                                                   int nrows, int ncols, int lrelu_a) {
    __shared__ __align__(16) _Float16 ldsA[16][IN_FEATS];
    int row0 = blockIdx.x * 16;
    int tid = threadIdx.x;
    for (int i = tid; i < 16 * IN_FEATS; i += 256) {
        int r = i >> 7, c = i & 127;
        int rg = row0 + r;
        float v = (rg < nrows) ? A[(size_t)rg * IN_FEATS + c] : 0.f;
        if (lrelu_a) v = v > 0.f ? v : v * 0.01f;
        ldsA[r][c] = (_Float16)v;
    }
    __syncthreads();
    int wave = tid >> 5, lane = tid & 31;
    int nl = lane & 15, half = lane >> 4;

    // preload A operands for all 4 K-steps (shared across col tiles)
    v16h av[4];
    #pragma unroll
    for (int t = 0; t < 4; t++) {
        v8h alo = *(const v8h*)&ldsA[nl][t * 32 + half * 8];
        v8h ahi = *(const v8h*)&ldsA[nl][t * 32 + 16 + half * 8];
        av[t] = combine8(alo, ahi);
    }

    bool full = (row0 + 16 <= nrows);
    for (int ct = wave; ct < (ncols >> 4); ct += 8) {
        int col0 = ct * 16;
        const _Float16* bbase = &Bt[(size_t)(col0 + nl) * IN_FEATS + half * 8];
        v16h bv[4];
        #pragma unroll
        for (int t = 0; t < 4; t++) {
            v8h b0 = *(const v8h*)(bbase + t * 32);
            v8h b1 = *(const v8h*)(bbase + t * 32 + 16);
            bv[t] = combine8(b0, b1);
        }
        v8f acc = {};
        #pragma unroll
        for (int t = 0; t < 4; t++)
            acc = __builtin_amdgcn_wmma_f32_16x16x32_f16(false, av[t], false, bv[t],
                                                         (short)0, acc, false, false);
        float* cp = C + (size_t)(row0 + 8 * half) * ncols + col0 + nl;
        if (full) {
            #pragma unroll
            for (int r = 0; r < 8; r++) cp[(size_t)r * ncols] = acc[r];
        } else {
            #pragma unroll
            for (int r = 0; r < 8; r++)
                if (row0 + r + 8 * half < nrows) cp[(size_t)r * ncols] = acc[r];
        }
    }
}

// ---------- conv1 attention scalars: a_src/a_dst [N,8] ----------
__global__ void k_attn1(const float* __restrict__ h, const float* __restrict__ att_src,
                        const float* __restrict__ att_dst, float* __restrict__ asrc,
                        float* __restrict__ adst, int Nn) {
    int i = blockIdx.x * blockDim.x + threadIdx.x;
    if (i >= Nn * HEADS) return;
    int node = i >> 3, hh = i & 7;
    const float* hp = h + (size_t)node * IN_FEATS + hh * HID;
    float s0 = 0.f, s1 = 0.f;
    #pragma unroll
    for (int c = 0; c < HID; c++) {
        float v = hp[c];
        s0 += v * att_src[hh * HID + c];
        s1 += v * att_dst[hh * HID + c];
    }
    asrc[i] = s0; adst[i] = s1;
}

// ---------- conv2 attention scalars: [N] ----------
__global__ void k_attn2(const float* __restrict__ h, const float* __restrict__ att_src,
                        const float* __restrict__ att_dst, float* __restrict__ asrc,
                        float* __restrict__ adst, int Nn) {
    int node = blockIdx.x * blockDim.x + threadIdx.x;
    if (node >= Nn) return;
    const float* hp = h + (size_t)node * IN_FEATS;
    float s0 = 0.f, s1 = 0.f;
    #pragma unroll 4
    for (int c = 0; c < IN_FEATS; c++) {
        float v = hp[c];
        s0 += v * att_src[c];
        s1 += v * att_dst[c];
    }
    asrc[node] = s0; adst[node] = s1;
}

__device__ __forceinline__ void edge_sd(const int* ei, int E, int e, int& s, int& d) {
    if (e < E) { s = ei[e]; d = ei[E + e]; } else { s = d = e - E; }
}

// ---------- conv1 edge passes ----------
__global__ void k_edge_max1(const int* __restrict__ ei, int E, int Nn,
                            const float* __restrict__ asrc, const float* __restrict__ adst,
                            unsigned* __restrict__ emax) {
    int e = blockIdx.x * blockDim.x + threadIdx.x;
    if (e >= E + Nn) return;
    int s, d; edge_sd(ei, E, e, s, d);
    #pragma unroll
    for (int hh = 0; hh < HEADS; hh++) {
        float x = lrelu(asrc[s * HEADS + hh] + adst[d * HEADS + hh], 0.2f);
        atomicMax(&emax[d * HEADS + hh], enc_f(x));
    }
}

__global__ void k_edge_expsum1(const int* __restrict__ ei, int E, int Nn,
                               const float* __restrict__ asrc, const float* __restrict__ adst,
                               const unsigned* __restrict__ emax, float* __restrict__ denom) {
    int e = blockIdx.x * blockDim.x + threadIdx.x;
    if (e >= E + Nn) return;
    int s, d; edge_sd(ei, E, e, s, d);
    #pragma unroll
    for (int hh = 0; hh < HEADS; hh++) {
        float x = lrelu(asrc[s * HEADS + hh] + adst[d * HEADS + hh], 0.2f);
        float w = __expf(x - dec_f(emax[d * HEADS + hh]));
        atomicAdd(&denom[d * HEADS + hh], w);
    }
}

__global__ void k_edge_aggr1(const int* __restrict__ ei, int E, int Nn,
                             const float* __restrict__ asrc, const float* __restrict__ adst,
                             const unsigned* __restrict__ emax, const float* __restrict__ denom,
                             const float* __restrict__ h, float* __restrict__ out) {
    int i = blockIdx.x * blockDim.x + threadIdx.x;
    if (i >= (E + Nn) * HEADS) return;
    int e = i >> 3, hh = i & 7;
    int s, d; edge_sd(ei, E, e, s, d);
    float x = lrelu(asrc[s * HEADS + hh] + adst[d * HEADS + hh], 0.2f);
    float alpha = __expf(x - dec_f(emax[d * HEADS + hh])) / (denom[d * HEADS + hh] + 1e-16f);
    const float* hp = h + (size_t)s * IN_FEATS + hh * HID;
    float* op = out + (size_t)d * IN_FEATS + hh * HID;
    #pragma unroll
    for (int c = 0; c < HID; c++) atomicAdd(&op[c], hp[c] * alpha);
}

// ---------- conv2 edge passes (heads=1) ----------
__global__ void k_edge_max2(const int* __restrict__ ei, int E, int Nn,
                            const float* __restrict__ asrc, const float* __restrict__ adst,
                            unsigned* __restrict__ emax) {
    int e = blockIdx.x * blockDim.x + threadIdx.x;
    if (e >= E + Nn) return;
    int s, d; edge_sd(ei, E, e, s, d);
    float x = lrelu(asrc[s] + adst[d], 0.2f);
    atomicMax(&emax[d], enc_f(x));
}

__global__ void k_edge_expsum2(const int* __restrict__ ei, int E, int Nn,
                               const float* __restrict__ asrc, const float* __restrict__ adst,
                               const unsigned* __restrict__ emax, float* __restrict__ denom) {
    int e = blockIdx.x * blockDim.x + threadIdx.x;
    if (e >= E + Nn) return;
    int s, d; edge_sd(ei, E, e, s, d);
    float x = lrelu(asrc[s] + adst[d], 0.2f);
    atomicAdd(&denom[d], __expf(x - dec_f(emax[d])));
}

__global__ void k_edge_aggr2(const int* __restrict__ ei, int E, int Nn,
                             const float* __restrict__ asrc, const float* __restrict__ adst,
                             const unsigned* __restrict__ emax, const float* __restrict__ denom,
                             const float* __restrict__ h, float* __restrict__ out) {
    int i = blockIdx.x * blockDim.x + threadIdx.x;
    if (i >= (E + Nn) * 32) return;
    int e = i >> 5, q = i & 31;
    int s, d; edge_sd(ei, E, e, s, d);
    float x = lrelu(asrc[s] + adst[d], 0.2f);
    float alpha = __expf(x - dec_f(emax[d])) / (denom[d] + 1e-16f);
    const float4 v = *((const float4*)(h + (size_t)s * IN_FEATS) + q);
    float* op = out + (size_t)d * IN_FEATS + q * 4;
    atomicAdd(op + 0, v.x * alpha);
    atomicAdd(op + 1, v.y * alpha);
    atomicAdd(op + 2, v.z * alpha);
    atomicAdd(op + 3, v.w * alpha);
}

// ---------- fused GEMM3 (H[N,128] @ g[128,512]) + cosine epilogue -> out [N,8] ----------
__global__ __launch_bounds__(256) void k_gemm3_cos(const float* __restrict__ H,
                                                   const _Float16* __restrict__ Gt,
                                                   const float* __restrict__ mu,
                                                   float* __restrict__ out, int Nn) {
    __shared__ __align__(16) _Float16 ldsA[16][IN_FEATS];
    __shared__ float ldsT[8][16][MDIM + 2];
    int row0 = blockIdx.x * 16;
    int tid = threadIdx.x;
    for (int i = tid; i < 16 * IN_FEATS; i += 256) {
        int r = i >> 7, c = i & 127;
        int rg = row0 + r;
        ldsA[r][c] = (_Float16)((rg < Nn) ? H[(size_t)rg * IN_FEATS + c] : 0.f);
    }
    __syncthreads();
    int wave = tid >> 5, lane = tid & 31;  // wave == k-group index (0..7)
    int nl = lane & 15, half = lane >> 4;

    v16h av[4];
    #pragma unroll
    for (int t = 0; t < 4; t++) {
        v8h alo = *(const v8h*)&ldsA[nl][t * 32 + half * 8];
        v8h ahi = *(const v8h*)&ldsA[nl][t * 32 + 16 + half * 8];
        av[t] = combine8(alo, ahi);
    }

    #pragma unroll
    for (int nt = 0; nt < 4; nt++) {
        int col0 = wave * MDIM + nt * 16;
        const _Float16* bbase = &Gt[(size_t)(col0 + nl) * IN_FEATS + half * 8];
        v16h bv[4];
        #pragma unroll
        for (int t = 0; t < 4; t++) {
            v8h b0 = *(const v8h*)(bbase + t * 32);
            v8h b1 = *(const v8h*)(bbase + t * 32 + 16);
            bv[t] = combine8(b0, b1);
        }
        v8f acc = {};
        #pragma unroll
        for (int t = 0; t < 4; t++)
            acc = __builtin_amdgcn_wmma_f32_16x16x32_f16(false, av[t], false, bv[t],
                                                         (short)0, acc, false, false);
        #pragma unroll
        for (int r = 0; r < 8; r++) ldsT[wave][r + 8 * half][nt * 16 + nl] = acc[r];
    }
    __syncthreads();
    // ||mu[k]||
    float mn = 0.f;
    #pragma unroll 8
    for (int m = 0; m < MDIM; m++) { float mv = mu[wave * MDIM + m]; mn += mv * mv; }
    mn = fmaxf(sqrtf(mn), 1e-8f);
    int row = lane >> 1, hp = lane & 1;
    float num = 0.f, sq = 0.f;
    for (int m = hp * 32; m < hp * 32 + 32; m++) {
        float v = ldsT[wave][row][m];
        num += mu[wave * MDIM + m] * v;
        sq += v * v;
    }
    num += __shfl_xor(num, 1, 32);
    sq  += __shfl_xor(sq, 1, 32);
    if (hp == 0 && (row0 + row) < Nn) {
        float dn = mn * fmaxf(sqrtf(sq), 1e-8f);
        out[(size_t)(row0 + row) * KTOT + wave] = num / dn;
    }
}

// ---------- launch ----------
extern "C" void kernel_launch(void* const* d_in, const int* in_sizes, int n_in,
                              void* d_out, int out_size, void* d_ws, size_t ws_size,
                              hipStream_t stream) {
    const float* x        = (const float*)d_in[0];
    const int*   ei       = (const int*)d_in[1];
    const float* W1       = (const float*)d_in[2];
    const float* att_src1 = (const float*)d_in[3];
    const float* att_dst1 = (const float*)d_in[4];
    const float* W2       = (const float*)d_in[6];
    const float* att_src2 = (const float*)d_in[7];
    const float* att_dst2 = (const float*)d_in[8];
    const float* g        = (const float*)d_in[10];
    const float* mu       = (const float*)d_in[11];
    float* out = (float*)d_out;

    int Nn = in_sizes[0] / IN_FEATS;
    int E = in_sizes[1] / 2;
    int Etot = E + Nn;

    char* ws = (char*)d_ws;
    size_t off = 0;
    auto alloc = [&](size_t bytes) -> void* {
        void* p = ws + off;
        off += (bytes + 255) & ~(size_t)255;
        return p;
    };
    float*     bufA  = (float*)alloc((size_t)Nn * IN_FEATS * 4);   // h1, then h2
    float*     bufB  = (float*)alloc((size_t)Nn * IN_FEATS * 4);   // out1, then out2
    float*     asrc1 = (float*)alloc((size_t)Nn * HEADS * 4);
    float*     adst1 = (float*)alloc((size_t)Nn * HEADS * 4);
    unsigned*  emax1 = (unsigned*)alloc((size_t)Nn * HEADS * 4);
    float*     den1  = (float*)alloc((size_t)Nn * HEADS * 4);
    float*     asrc2 = (float*)alloc((size_t)Nn * 4);
    float*     adst2 = (float*)alloc((size_t)Nn * 4);
    unsigned*  emax2 = (unsigned*)alloc((size_t)Nn * 4);
    float*     den2  = (float*)alloc((size_t)Nn * 4);
    _Float16*  W1t   = (_Float16*)alloc((size_t)IN_FEATS * IN_FEATS * 2);
    _Float16*  W2t   = (_Float16*)alloc((size_t)IN_FEATS * IN_FEATS * 2);
    _Float16*  Gt    = (_Float16*)alloc((size_t)IN_FEATS * GCOLS * 2);

    hipMemsetAsync(bufB, 0, (size_t)Nn * IN_FEATS * 4, stream);
    hipMemsetAsync(emax1, 0, (size_t)Nn * HEADS * 4, stream);  // enc key 0 < enc(-inf)
    hipMemsetAsync(den1, 0, (size_t)Nn * HEADS * 4, stream);
    hipMemsetAsync(emax2, 0, (size_t)Nn * 4, stream);
    hipMemsetAsync(den2, 0, (size_t)Nn * 4, stream);

    k_transpose_f16<<<(IN_FEATS * IN_FEATS + 255) / 256, 256, 0, stream>>>(W1, W1t, IN_FEATS, IN_FEATS);
    k_transpose_f16<<<(IN_FEATS * IN_FEATS + 255) / 256, 256, 0, stream>>>(W2, W2t, IN_FEATS, IN_FEATS);
    k_transpose_f16<<<(IN_FEATS * GCOLS + 255) / 256, 256, 0, stream>>>(g, Gt, IN_FEATS, GCOLS);

    int rowBlocks = (Nn + 15) / 16;
    // conv1
    k_gemm_wmma<<<rowBlocks, 256, 0, stream>>>(x, W1t, bufA, Nn, IN_FEATS, 0);
    k_attn1<<<(Nn * HEADS + 255) / 256, 256, 0, stream>>>(bufA, att_src1, att_dst1, asrc1, adst1, Nn);
    k_edge_max1<<<(Etot + 255) / 256, 256, 0, stream>>>(ei, E, Nn, asrc1, adst1, emax1);
    k_edge_expsum1<<<(Etot + 255) / 256, 256, 0, stream>>>(ei, E, Nn, asrc1, adst1, emax1, den1);
    k_edge_aggr1<<<(Etot * HEADS + 255) / 256, 256, 0, stream>>>(ei, E, Nn, asrc1, adst1, emax1, den1, bufA, bufB);
    // conv2 (leaky 0.01 fused into GEMM2's A load)
    k_gemm_wmma<<<rowBlocks, 256, 0, stream>>>(bufB, W2t, bufA, Nn, IN_FEATS, 1);
    k_attn2<<<(Nn + 255) / 256, 256, 0, stream>>>(bufA, att_src2, att_dst2, asrc2, adst2, Nn);
    k_edge_max2<<<(Etot + 255) / 256, 256, 0, stream>>>(ei, E, Nn, asrc2, adst2, emax2);
    k_edge_expsum2<<<(Etot + 255) / 256, 256, 0, stream>>>(ei, E, Nn, asrc2, adst2, emax2, den2);
    hipMemsetAsync(bufB, 0, (size_t)Nn * IN_FEATS * 4, stream);
    k_edge_aggr2<<<((Etot * 32) + 255) / 256, 256, 0, stream>>>(ei, E, Nn, asrc2, adst2, emax2, den2, bufA, bufB);
    // projection + cosine
    k_gemm3_cos<<<rowBlocks, 256, 0, stream>>>(bufB, Gt, mu, out, Nn);

    (void)n_in; (void)out_size; (void)ws_size;
}
